// CausalSelfAttention_28037546508998
// MI455X (gfx1250) — matrix operand order repaired and checked by reading, not compiled, not run on hardware
//
#include <hip/hip_runtime.h>

typedef unsigned short u16;
typedef __attribute__((ext_vector_type(16))) __bf16 v16bf;
typedef __attribute__((ext_vector_type(8)))  float  v8f;
typedef __attribute__((vector_size(4 * sizeof(int)))) int i32x4;

#if defined(__has_builtin)
#if __has_builtin(__builtin_amdgcn_global_load_async_to_lds_b128) && \
    __has_builtin(__builtin_amdgcn_s_wait_asynccnt)
#define USE_ASYNC_LDS 1
#else
#define USE_ASYNC_LDS 0
#endif
#else
#define USE_ASYNC_LDS 0
#endif

union BFr {
    v16bf v;
    uint4 q[2];
    u16   e[16];
};

__device__ __forceinline__ u16 f2bf(float f) {
    unsigned x = __float_as_uint(f);
    unsigned r = (x + 0x7FFFu + ((x >> 16) & 1u)) >> 16;   // RNE
    return (u16)r;
}

__device__ __forceinline__ v8f wmma_bf16(const BFr& a, const BFr& b, v8f c) {
    return __builtin_amdgcn_wmma_f32_16x16x32_bf16(
        /*neg_a=*/false, a.v, /*neg_b=*/false, b.v,
        /*c_mod=*/(short)0, c, /*reuse_a=*/false, /*reuse_b=*/false);
}

// ---- async global->LDS copy (GLOBAL_LOAD_ASYNC_TO_LDS_B128, ASYNCcnt) ----
__device__ __forceinline__ void cp_b128(void* lds, const void* g) {
#if USE_ASYNC_LDS
    __builtin_amdgcn_global_load_async_to_lds_b128(
        (__attribute__((address_space(1))) i32x4*)(void*)g,
        (__attribute__((address_space(3))) i32x4*)lds,
        0, 0);
#else
    *(uint4*)lds = *(const uint4*)g;
#endif
}
__device__ __forceinline__ void cp_wait() {
#if USE_ASYNC_LDS
    __builtin_amdgcn_s_wait_asynccnt(0);
#endif
}

// ---------------------------------------------------------------- conversions
__global__ void cvt_f32_bf16(const float* __restrict__ in, u16* __restrict__ out, int n) {
    int i = blockIdx.x * 256 + threadIdx.x;
    if (i < n) out[i] = f2bf(in[i]);
}

// ---------------------------------------------------------------- GEMM
// C[M,N] = A[M,K] @ B[K,N] + bias.  A: f32 (MODE 0, x) or bf16 (MODE 1, att-out).
// MODE 0 epilogue: scatter bf16 to Q/K/V [B,H,T,HD].  MODE 1: f32 to d_out.
// WG = 128 threads (4 waves). WG tile 64(M) x 256(N); per-wave 64x64 -> 16 acc tiles.
template <int MODE>
__global__ __launch_bounds__(128) void gemm_kernel(
    const void* __restrict__ Ain, const u16* __restrict__ Bw,
    const float* __restrict__ bias,
    u16* __restrict__ qo, u16* __restrict__ ko, u16* __restrict__ vo,
    float* __restrict__ co, int M, int N, int K)
{
    __shared__ __align__(16) u16 lA[64][40];    // [m][k], 80B rows (16B aligned)
    __shared__ __align__(16) u16 lBT[256][40];  // [n][k] (transposed), 80B rows

    const int tid  = threadIdx.x;
    const int wave = tid >> 5;
    const int lane = tid & 31;
    const int hl   = lane >> 4;     // which 16-lane half
    const int l15  = lane & 15;
    const int mbase = blockIdx.x * 64;
    const int nbase = blockIdx.y * 256;

    v8f acc[4][4] = {};

    for (int k0 = 0; k0 < K; k0 += 32) {
        // ---- stage A tile 64x32 (fuse f32->bf16 for MODE 0; async copy MODE 1) ----
        if (MODE == 0) {
            const float* A = (const float*)Ain;
            for (int c = tid; c < 256; c += 128) {
                int r = c >> 2, kp = (c & 3) * 8;
                const float* p = A + (size_t)(mbase + r) * K + k0 + kp;
                float4 a0 = *(const float4*)p;
                float4 a1 = *(const float4*)(p + 4);
                uint4 pk;
                pk.x = (unsigned)f2bf(a0.x) | ((unsigned)f2bf(a0.y) << 16);
                pk.y = (unsigned)f2bf(a0.z) | ((unsigned)f2bf(a0.w) << 16);
                pk.z = (unsigned)f2bf(a1.x) | ((unsigned)f2bf(a1.y) << 16);
                pk.w = (unsigned)f2bf(a1.z) | ((unsigned)f2bf(a1.w) << 16);
                *(uint4*)&lA[r][kp] = pk;
                if (k0 + 32 < K) __builtin_prefetch(p + 32, 0, 1);
            }
        } else {
            const u16* A = (const u16*)Ain;
            for (int c = tid; c < 256; c += 128) {
                int r = c >> 2, kp = (c & 3) * 8;
                const u16* p = A + (size_t)(mbase + r) * K + k0 + kp;
                cp_b128(&lA[r][kp], p);                      // async -> LDS
                if (k0 + 32 < K) __builtin_prefetch(p + 32, 0, 1);
            }
        }
        // ---- stage B tile 32x256, transposed into lBT[n][k] ----
        for (int c = tid; c < 1024; c += 128) {
            int kk = c >> 5, n8 = (c & 31) * 8;
            const u16* p = Bw + (size_t)(k0 + kk) * N + nbase + n8;
            uint4 vb = *(const uint4*)p;
            const u16* e = (const u16*)&vb;
#pragma unroll
            for (int j = 0; j < 8; ++j) lBT[n8 + j][kk] = e[j];
            if (k0 + 32 < K) __builtin_prefetch(p + (size_t)32 * N, 0, 1);
        }
        cp_wait();
        __syncthreads();

        // ---- fragments ----
        BFr af[4], bfr[4];
#pragma unroll
        for (int mt = 0; mt < 4; ++mt) {
            int r = mt * 16 + l15;                       // A rows: lane&15 (both halves)
            af[mt].q[0] = *(const uint4*)&lA[r][hl * 8];        // K = hl*8 + 0..7
            af[mt].q[1] = *(const uint4*)&lA[r][16 + hl * 8];   // K = 16 + hl*8 + 0..7
        }
#pragma unroll
        for (int nt = 0; nt < 4; ++nt) {
            int n = wave * 64 + nt * 16 + l15;           // B col: lane&15
            bfr[nt].q[0] = *(const uint4*)&lBT[n][hl * 8];      // K = hl*8 + 0..7
            bfr[nt].q[1] = *(const uint4*)&lBT[n][16 + hl * 8]; // K = 16 + hl*8 + 0..7
        }
#pragma unroll
        for (int mt = 0; mt < 4; ++mt)
#pragma unroll
            for (int nt = 0; nt < 4; ++nt)
                acc[mt][nt] = wmma_bf16(af[mt], bfr[nt], acc[mt][nt]);
        __syncthreads();
    }

    // ---- epilogue ----
#pragma unroll
    for (int mt = 0; mt < 4; ++mt) {
#pragma unroll
        for (int nt = 0; nt < 4; ++nt) {
            int col = nbase + wave * 64 + nt * 16 + l15;
            float bv = bias[col];
#pragma unroll
            for (int j = 0; j < 8; ++j) {
                int row = mbase + mt * 16 + hl * 8 + j;  // C layout: M = hl*8 + j
                float val = acc[mt][nt][j] + bv;
                if (MODE == 0) {
                    int which = col >> 10;               // 0=q 1=k 2=v
                    int d = col & 1023;
                    int head = d >> 6, hd = d & 63;
                    int b = row >> 10, t = row & 1023;
                    size_t idx = (((size_t)b * 16 + head) * 1024 + t) * 64 + hd;
                    u16* dst = (which == 0) ? qo : (which == 1) ? ko : vo;
                    dst[idx] = f2bf(val);
                } else {
                    co[(size_t)row * N + col] = val;
                }
            }
        }
    }
}

// ---------------------------------------------------------------- attention
// grid (B*H=128, T/64=16), 128 threads (4 waves), wave owns 16 query rows.
__global__ __launch_bounds__(128) void attn_kernel(
    const u16* __restrict__ Q, const u16* __restrict__ Km,
    const u16* __restrict__ V, u16* __restrict__ Oout)
{
    __shared__ __align__(16) u16 lK[64][72];        // [key][hd]   144B rows
    __shared__ __align__(16) u16 lVT[64][72];       // [hd][key]   transposed
    __shared__ __align__(16) u16 lP[4][16][72];     // per-wave P relayout

    const int tid  = threadIdx.x;
    const int wave = tid >> 5;
    const int lane = tid & 31;
    const int hl   = lane >> 4;
    const int l15  = lane & 15;
    const int bh   = blockIdx.x;       // b*16 + h
    const int qb   = blockIdx.y;       // query block of 64
    const size_t bhT = (size_t)bh * 1024;

    // Q A-fragments (2 K-chunks of 32 over HD=64), kept in registers
    const int qrow = qb * 64 + wave * 16 + l15;
    BFr qf[2];
#pragma unroll
    for (int kc = 0; kc < 2; ++kc) {
        const u16* p = Q + (bhT + qrow) * 64 + kc * 32;
        qf[kc].q[0] = *(const uint4*)(p + hl * 8);
        qf[kc].q[1] = *(const uint4*)(p + 16 + hl * 8);
    }

    float mrow[8], lrow[8];
    v8f o[4] = {};
#pragma unroll
    for (int j = 0; j < 8; ++j) { mrow[j] = -3.0e38f; lrow[j] = 0.0f; }

    const int nkv = qb + 1;            // causal: kv blocks 0..qb
    for (int jb = 0; jb < nkv; ++jb) {
        // ---- stage K (async, row-major) and V (transposed via VALU) ----
        for (int c = tid; c < 512; c += 128) {
            int key = c >> 3, hp = (c & 7) * 8;
            size_t g = (bhT + (size_t)jb * 64 + key) * 64 + hp;
            cp_b128(&lK[key][hp], Km + g);               // async -> LDS
            uint4 vv = *(const uint4*)(V + g);
            const u16* e = (const u16*)&vv;
#pragma unroll
            for (int t = 0; t < 8; ++t) lVT[hp + t][key] = e[t];
            if (jb + 1 < nkv) {
                __builtin_prefetch(Km + g + 64 * 64, 0, 1);
                __builtin_prefetch(V + g + 64 * 64, 0, 1);
            }
        }
        cp_wait();
        __syncthreads();

        // ---- S = Q @ K^T  (4 key tiles x 2 K-chunks = 8 WMMA) ----
        v8f s[4] = {};
#pragma unroll
        for (int kc = 0; kc < 2; ++kc) {
#pragma unroll
            for (int nt = 0; nt < 4; ++nt) {
                BFr kf;
                int key = nt * 16 + l15;
                kf.q[0] = *(const uint4*)&lK[key][kc * 32 + hl * 8];
                kf.q[1] = *(const uint4*)&lK[key][kc * 32 + 16 + hl * 8];
                s[nt] = wmma_bf16(qf[kc], kf, s[nt]);
            }
        }

        // ---- scale, causal mask, online softmax ----
#pragma unroll
        for (int j = 0; j < 8; ++j) {
            int qg = qb * 64 + wave * 16 + hl * 8 + j;
            float mx = -3.0e38f;
#pragma unroll
            for (int nt = 0; nt < 4; ++nt) {
                int kg = jb * 64 + nt * 16 + l15;
                float sv = s[nt][j] * 0.125f;            // 1/sqrt(64)
                sv = (kg > qg) ? -3.0e38f : sv;
                s[nt][j] = sv;
                mx = fmaxf(mx, sv);
            }
            for (int off = 1; off < 16; off <<= 1)
                mx = fmaxf(mx, __shfl_xor(mx, off, 32));
            float nm = fmaxf(mrow[j], mx);
            float alpha = __expf(mrow[j] - nm);
            float rs = 0.0f;
#pragma unroll
            for (int nt = 0; nt < 4; ++nt) {
                float p = __expf(s[nt][j] - nm);
                s[nt][j] = p;
                rs += p;
            }
            for (int off = 1; off < 16; off <<= 1)
                rs += __shfl_xor(rs, off, 32);
            mrow[j] = nm;
            lrow[j] = lrow[j] * alpha + rs;
#pragma unroll
            for (int nt = 0; nt < 4; ++nt) o[nt][j] *= alpha;
        }

        // ---- relayout P via per-wave LDS (C-layout -> A-fragment) ----
#pragma unroll
        for (int nt = 0; nt < 4; ++nt)
#pragma unroll
            for (int j = 0; j < 8; ++j)
                lP[wave][hl * 8 + j][nt * 16 + l15] = f2bf(s[nt][j]);

        // ---- O += P @ V  (2 key-chunks x 4 hd tiles = 8 WMMA) ----
#pragma unroll
        for (int kc = 0; kc < 2; ++kc) {
            BFr pf;
            pf.q[0] = *(const uint4*)&lP[wave][l15][kc * 32 + hl * 8];
            pf.q[1] = *(const uint4*)&lP[wave][l15][kc * 32 + 16 + hl * 8];
#pragma unroll
            for (int nt = 0; nt < 4; ++nt) {
                BFr vf;
                int hd = nt * 16 + l15;
                vf.q[0] = *(const uint4*)&lVT[hd][kc * 32 + hl * 8];
                vf.q[1] = *(const uint4*)&lVT[hd][kc * 32 + 16 + hl * 8];
                o[nt] = wmma_bf16(pf, vf, o[nt]);
            }
        }
        __syncthreads();
    }

    // ---- normalize + write att-out bf16 as [B*T, D] rows for proj GEMM ----
    const int b = bh >> 4, h = bh & 15;
#pragma unroll
    for (int nt = 0; nt < 4; ++nt) {
#pragma unroll
        for (int j = 0; j < 8; ++j) {
            int t = qb * 64 + wave * 16 + hl * 8 + j;
            int col = h * 64 + nt * 16 + l15;
            Oout[((size_t)b * 1024 + t) * 1024 + col] = f2bf(o[nt][j] / lrow[j]);
        }
    }
}

// ---------------------------------------------------------------- launch
extern "C" void kernel_launch(void* const* d_in, const int* in_sizes, int n_in,
                              void* d_out, int out_size, void* d_ws, size_t ws_size,
                              hipStream_t stream) {
    const float* x      = (const float*)d_in[0];  // [8,1024,1024]
    const float* w_qkv  = (const float*)d_in[1];  // [1024,3072]
    const float* b_qkv  = (const float*)d_in[2];  // [3072]
    const float* w_proj = (const float*)d_in[3];  // [1024,1024]
    const float* b_proj = (const float*)d_in[4];  // [1024]
    float* out = (float*)d_out;                   // [8,1024,1024]

    char* ws = (char*)d_ws;
    size_t off = 0;
    auto carve = [&](size_t bytes) -> u16* {
        u16* p = (u16*)(ws + off);
        off += (bytes + 255) & ~(size_t)255;
        return p;
    };
    u16* wqkv_bf  = carve((size_t)1024 * 3072 * 2);
    u16* wproj_bf = carve((size_t)1024 * 1024 * 2);
    u16* q_bf     = carve((size_t)8 * 16 * 1024 * 64 * 2);
    u16* k_bf     = carve((size_t)8 * 16 * 1024 * 64 * 2);
    u16* v_bf     = carve((size_t)8 * 16 * 1024 * 64 * 2);
    u16* ao_bf    = carve((size_t)8 * 1024 * 1024 * 2);

    cvt_f32_bf16<<<(3145728 + 255) / 256, 256, 0, stream>>>(w_qkv, wqkv_bf, 3145728);
    cvt_f32_bf16<<<(1048576 + 255) / 256, 256, 0, stream>>>(w_proj, wproj_bf, 1048576);

    // QKV: [8192,1024] @ [1024,3072] -> scatter q/k/v
    gemm_kernel<0><<<dim3(128, 12), 128, 0, stream>>>(
        (const void*)x, wqkv_bf, b_qkv, q_bf, k_bf, v_bf, nullptr, 8192, 3072, 1024);

    // flash attention per (b,h) x query-block
    attn_kernel<<<dim3(128, 16), 128, 0, stream>>>(q_bf, k_bf, v_bf, ao_bf);

    // proj: [8192,1024] @ [1024,1024] + bias -> out f32
    gemm_kernel<1><<<dim3(128, 4), 128, 0, stream>>>(
        (const void*)ao_bf, wproj_bf, b_proj, nullptr, nullptr, nullptr, out, 8192, 1024, 1024);
}
